// EquivariantMPNNLayer_68324339745418
// MI455X (gfx1250) — matrix operand order repaired and checked by reading, not compiled, not run on hardware
//
#include <hip/hip_runtime.h>
#include <hip/hip_fp16.h>

// ---------------------------------------------------------------------------
// EquivariantMPNNLayer fused WMMA implementation for gfx1250 (MI455X, wave32)
//   Kernel A: per-edge frame-rotate + posMLP + msgMLP + atomic scatter-add
//   Kernel B: segment mean + update MLP
// All GEMMs via v_wmma_f32_16x16x32_f16 (f16 in, f32 accumulate).
// LDS rows padded (+8/+8 halves) so row-to-row stride advances 4 banks:
// 16 lanes reading 16 consecutive rows hit 16 distinct banks (no conflicts).
// ---------------------------------------------------------------------------

typedef __attribute__((ext_vector_type(16))) _Float16 v16h;
typedef __attribute__((ext_vector_type(8)))  _Float16 v8h;
typedef __attribute__((ext_vector_type(8)))  float    v8f;

#define HDIM 128

// Padded LDS row strides (in halves); all are 16-byte multiples.
#define WSTRIDE  136   // K=128 weight tiles (We2t/Wm2t/Wu1t/Wu2t)
#define W1STRIDE 264   // K=256 weight tile (Wm1t)
#define MSTRIDE  264   // per-wave msg buffer rows (256 data + 8 pad)
#define INSTRIDE 136   // kernel-B per-wave input tile rows (128 data + 8 pad)

static __device__ __forceinline__ float silu_f(float x) {
    return x / (1.0f + __expf(-x));
}

static __device__ __forceinline__ v8f wmma16(v16h a, v16h b, v8f c) {
    // 8 args: (neg_a, A, neg_b, B, c_mod, C, reuse_a, reuse_b)
    return __builtin_amdgcn_wmma_f32_16x16x32_f16(false, a, false, b,
                                                  (short)0, c, false, false);
}

// Load 16 halves as two aligned 8-half chunks at (p+o0) and (p+o1).
static __device__ __forceinline__ v16h ld_pair(const _Float16* p, int o0, int o1) {
    v8h lo = *(const v8h*)(p + o0);
    v8h hi = *(const v8h*)(p + o1);
    return __builtin_shufflevector(lo, hi, 0, 1, 2, 3, 4, 5, 6, 7,
                                           8, 9, 10, 11, 12, 13, 14, 15);
}

// LDS layout, kernel A (bytes)
#define A_WM1T   0        // [128][264] f16  = 67584
#define A_WE2T   67584    // [128][136] f16  = 34816
#define A_WM2T   102400   // [128][136] f16  = 34816
#define A_WE1    137216   // [3][128]  f32   = 1536
#define A_BE1    138752   // 512
#define A_BE2    139264   // 512
#define A_BM1    139776   // 512
#define A_BM2    140288   // 512
#define A_MSG    140800   // [8][16][264] f16 = 67584
#define A_LR     208384   // [8][16][3] f32   = 1536
#define A_J      209920   // [8][16] i32      = 512
#define SMEM_A   210432

// LDS layout, kernel B (bytes)
#define B_WU1T   0        // [128][136] f16 = 34816
#define B_WU2T   34816    // 34816
#define B_BU1    69632    // 512
#define B_BU2    70144    // 512
#define B_IN     70656    // [8][16][136] f16 = 34816
#define SMEM_B   105472

__global__ __launch_bounds__(256, 1)
void mpnn_edge_kernel(const float* __restrict__ node_emb,
                      const float* __restrict__ node_pos,
                      const float* __restrict__ grid_pos,
                      const int*   __restrict__ eidx,     // [2][E]
                      const float* __restrict__ frames,   // [B][3][3]
                      const int*   __restrict__ batch,
                      const float* __restrict__ We1, const float* __restrict__ be1,
                      const float* __restrict__ We2, const float* __restrict__ be2,
                      const float* __restrict__ Wm1, const float* __restrict__ bm1,
                      const float* __restrict__ Wm2, const float* __restrict__ bm2,
                      float* __restrict__ seg_sum, float* __restrict__ seg_cnt,
                      int E, int Nn)
{
    extern __shared__ char smem[];
    _Float16* sWm1t = (_Float16*)(smem + A_WM1T);
    _Float16* sWe2t = (_Float16*)(smem + A_WE2T);
    _Float16* sWm2t = (_Float16*)(smem + A_WM2T);
    float*    sWe1  = (float*)(smem + A_WE1);
    float*    sbe1  = (float*)(smem + A_BE1);
    float*    sbe2  = (float*)(smem + A_BE2);
    float*    sbm1  = (float*)(smem + A_BM1);
    float*    sbm2  = (float*)(smem + A_BM2);
    _Float16* smsg  = (_Float16*)(smem + A_MSG);
    float*    slr   = (float*)(smem + A_LR);
    int*      sj    = (int*)(smem + A_J);

    // --- stage weights transposed (N-major, padded rows) f16; biases f32 ---
    for (int idx = threadIdx.x; idx < 256 * 128; idx += 256) {
        int k = idx >> 7, n = idx & 127;           // Wm1: [256][128]
        sWm1t[n * W1STRIDE + k] = (_Float16)Wm1[idx];
    }
    for (int idx = threadIdx.x; idx < 128 * 128; idx += 256) {
        int k = idx >> 7, n = idx & 127;           // We2/Wm2: [128][128]
        sWe2t[n * WSTRIDE + k] = (_Float16)We2[idx];
        sWm2t[n * WSTRIDE + k] = (_Float16)Wm2[idx];
    }
    if (threadIdx.x < 128) {
        int c = threadIdx.x;
        sWe1[c] = We1[c]; sWe1[128 + c] = We1[128 + c]; sWe1[256 + c] = We1[256 + c];
        sbe1[c] = be1[c]; sbe2[c] = be2[c]; sbm1[c] = bm1[c]; sbm2[c] = bm2[c];
    }
    __syncthreads();

    const int wave = threadIdx.x >> 5;
    const int lane = threadIdx.x & 31;
    const int tile = (blockIdx.x * 8 + wave) * 16;

    _Float16* msg = smsg + wave * 16 * MSTRIDE;
    float*    lr  = slr  + wave * 16 * 3;
    int*      jv  = sj   + wave * 16;

    // --- per-edge meta: local_rel = frame @ (grid_pos[j-N] - node_pos[i]) ---
    if (lane < 16) {
        int e  = tile + lane;
        int ec = e < E ? e : E - 1;
        int i  = eidx[ec];
        int j  = eidx[E + ec];
        float rx = grid_pos[(j - Nn) * 3 + 0] - node_pos[i * 3 + 0];
        float ry = grid_pos[(j - Nn) * 3 + 1] - node_pos[i * 3 + 1];
        float rz = grid_pos[(j - Nn) * 3 + 2] - node_pos[i * 3 + 2];
        const float* F = frames + batch[i] * 9;
        lr[lane * 3 + 0] = F[0] * rx + F[1] * ry + F[2] * rz;
        lr[lane * 3 + 1] = F[3] * rx + F[4] * ry + F[5] * rz;
        lr[lane * 3 + 2] = F[6] * rx + F[7] * ry + F[8] * rz;
        jv[lane] = j;
    }

    // --- gather node embedding -> msg[:, 0:128] as f16 (2 lanes per row) ---
    {
        int m = lane >> 1, half = lane & 1;
        int e  = tile + m;
        int ec = e < E ? e : E - 1;
        int i  = eidx[ec];
        const float4* src = (const float4*)(node_emb + (size_t)i * HDIM + half * 64);
        _Float16* dst = msg + m * MSTRIDE + half * 64;
        #pragma unroll
        for (int c = 0; c < 16; ++c) {
            float4 v = src[c];
            dst[c * 4 + 0] = (_Float16)v.x; dst[c * 4 + 1] = (_Float16)v.y;
            dst[c * 4 + 2] = (_Float16)v.z; dst[c * 4 + 3] = (_Float16)v.w;
        }
    }

    // Fragment index constants (ISA 7.12.2, 16-bit A / B layouts, wave32):
    const int mrow = (lane & 16) ? 8 : 0;   // C/D: VGPR r -> M = mrow + r
    const int ncol = lane & 15;             // B/C/D column within 16-tile
    const int kb_a = (lane & 16) ? 8 : 0;   // A: lanes 16-31 start at K=8
    const int kb_b = (lane & 16) ? 8 : 0;   // B: lanes 16-31 start at K=8

    const float lr0 = lr[(lane & 15) * 3 + 0];
    const float lr1 = lr[(lane & 15) * 3 + 1];
    const float lr2 = lr[(lane & 15) * 3 + 2];

    v8f acc[8];

    // ======================= posMLP layer 2: h1 @ We2 =======================
    // h1[m][k] = silu(local_rel[m] . We1[:,k] + be1[k]) computed directly into
    // the A-fragment register pattern (K pattern: kb + (t&7) + (t>>3)*16).
    #pragma unroll
    for (int nt = 0; nt < 8; ++nt) { v8f z = {}; acc[nt] = z; }
    #pragma unroll
    for (int ks = 0; ks < 4; ++ks) {
        v16h a;
        #pragma unroll
        for (int t = 0; t < 16; ++t) {
            int k = ks * 32 + kb_a + (t & 7) + ((t >> 3) << 4);
            float x = lr0 * sWe1[k] + lr1 * sWe1[128 + k] + lr2 * sWe1[256 + k] + sbe1[k];
            a[t] = (_Float16)silu_f(x);
        }
        #pragma unroll
        for (int nt = 0; nt < 8; ++nt) {
            v16h b = ld_pair(sWe2t + (nt * 16 + ncol) * WSTRIDE + ks * 32 + kb_b, 0, 16);
            acc[nt] = wmma16(a, b, acc[nt]);
        }
    }
    // pos_emb = acc + be2 -> msg[:, 128:256] f16 (C-layout -> row-major store)
    #pragma unroll
    for (int nt = 0; nt < 8; ++nt) {
        float bb = sbe2[nt * 16 + ncol];
        #pragma unroll
        for (int r = 0; r < 8; ++r)
            msg[(mrow + r) * MSTRIDE + 128 + nt * 16 + ncol] = (_Float16)(acc[nt][r] + bb);
    }

    // ===================== msgMLP layer 1: msg @ Wm1 (K=256) =================
    #pragma unroll
    for (int nt = 0; nt < 8; ++nt) { v8f z = {}; acc[nt] = z; }
    #pragma unroll
    for (int ks = 0; ks < 8; ++ks) {
        v16h a = ld_pair(msg + (lane & 15) * MSTRIDE + ks * 32 + kb_a, 0, 16);
        #pragma unroll
        for (int nt = 0; nt < 8; ++nt) {
            v16h b = ld_pair(sWm1t + (nt * 16 + ncol) * W1STRIDE + ks * 32 + kb_b, 0, 16);
            acc[nt] = wmma16(a, b, acc[nt]);
        }
    }
    // hidden = silu(acc + bm1) -> overwrite msg[:, 0:128]
    #pragma unroll
    for (int nt = 0; nt < 8; ++nt) {
        float bb = sbm1[nt * 16 + ncol];
        #pragma unroll
        for (int r = 0; r < 8; ++r)
            msg[(mrow + r) * MSTRIDE + nt * 16 + ncol] = (_Float16)silu_f(acc[nt][r] + bb);
    }

    // ===================== msgMLP layer 2: hidden @ Wm2 ======================
    #pragma unroll
    for (int nt = 0; nt < 8; ++nt) { v8f z = {}; acc[nt] = z; }
    #pragma unroll
    for (int ks = 0; ks < 4; ++ks) {
        v16h a = ld_pair(msg + (lane & 15) * MSTRIDE + ks * 32 + kb_a, 0, 16);
        #pragma unroll
        for (int nt = 0; nt < 8; ++nt) {
            v16h b = ld_pair(sWm2t + (nt * 16 + ncol) * WSTRIDE + ks * 32 + kb_b, 0, 16);
            acc[nt] = wmma16(a, b, acc[nt]);
        }
    }

    // ================== scatter-add (L2-resident f32 atomics) ===============
    #pragma unroll
    for (int r = 0; r < 8; ++r) {
        int m = mrow + r;
        int e = tile + m;
        if (e < E) {
            float* dst = seg_sum + (size_t)jv[m] * HDIM;
            #pragma unroll
            for (int nt = 0; nt < 8; ++nt)
                unsafeAtomicAdd(dst + nt * 16 + ncol, acc[nt][r] + sbm2[nt * 16 + ncol]);
        }
    }
    if (lane < 16) {
        int e = tile + lane;
        if (e < E) unsafeAtomicAdd(seg_cnt + jv[lane], 1.0f);
    }
}

__global__ __launch_bounds__(256, 2)
void mpnn_update_kernel(const float* __restrict__ seg_sum,
                        const float* __restrict__ seg_cnt,
                        const float* __restrict__ Wu1, const float* __restrict__ bu1,
                        const float* __restrict__ Wu2, const float* __restrict__ bu2,
                        float* __restrict__ out, int NS)
{
    extern __shared__ char smem[];
    _Float16* sWu1t = (_Float16*)(smem + B_WU1T);
    _Float16* sWu2t = (_Float16*)(smem + B_WU2T);
    float*    sbu1  = (float*)(smem + B_BU1);
    float*    sbu2  = (float*)(smem + B_BU2);
    _Float16* sin   = (_Float16*)(smem + B_IN);

    for (int idx = threadIdx.x; idx < 128 * 128; idx += 256) {
        int k = idx >> 7, n = idx & 127;
        sWu1t[n * WSTRIDE + k] = (_Float16)Wu1[idx];
        sWu2t[n * WSTRIDE + k] = (_Float16)Wu2[idx];
    }
    if (threadIdx.x < 128) {
        sbu1[threadIdx.x] = bu1[threadIdx.x];
        sbu2[threadIdx.x] = bu2[threadIdx.x];
    }
    __syncthreads();

    const int wave = threadIdx.x >> 5;
    const int lane = threadIdx.x & 31;
    const int base = (blockIdx.x * 8 + wave) * 16;
    _Float16* in = sin + wave * 16 * INSTRIDE;

    // --- mean = seg_sum / max(cnt,1) -> f16 tile (2 lanes per row) ---
    {
        int m = lane >> 1, half = lane & 1;
        int row = base + m;
        int rc  = row < NS ? row : NS - 1;
        float inv = 1.0f / fmaxf(seg_cnt[rc], 1.0f);
        const float4* src = (const float4*)(seg_sum + (size_t)rc * HDIM + half * 64);
        _Float16* dst = in + m * INSTRIDE + half * 64;
        #pragma unroll
        for (int c = 0; c < 16; ++c) {
            float4 v = src[c];
            dst[c * 4 + 0] = (_Float16)(v.x * inv); dst[c * 4 + 1] = (_Float16)(v.y * inv);
            dst[c * 4 + 2] = (_Float16)(v.z * inv); dst[c * 4 + 3] = (_Float16)(v.w * inv);
        }
    }

    const int mrow = (lane & 16) ? 8 : 0;
    const int ncol = lane & 15;
    const int kb_a = (lane & 16) ? 8 : 0;
    const int kb_b = (lane & 16) ? 8 : 0;

    v8f acc[8];

    // --- layer u1: silu(mean @ Wu1 + bu1) ---
    #pragma unroll
    for (int nt = 0; nt < 8; ++nt) { v8f z = {}; acc[nt] = z; }
    #pragma unroll
    for (int ks = 0; ks < 4; ++ks) {
        v16h a = ld_pair(in + (lane & 15) * INSTRIDE + ks * 32 + kb_a, 0, 16);
        #pragma unroll
        for (int nt = 0; nt < 8; ++nt) {
            v16h b = ld_pair(sWu1t + (nt * 16 + ncol) * WSTRIDE + ks * 32 + kb_b, 0, 16);
            acc[nt] = wmma16(a, b, acc[nt]);
        }
    }
    #pragma unroll
    for (int nt = 0; nt < 8; ++nt) {
        float bb = sbu1[nt * 16 + ncol];
        #pragma unroll
        for (int r = 0; r < 8; ++r)
            in[(mrow + r) * INSTRIDE + nt * 16 + ncol] = (_Float16)silu_f(acc[nt][r] + bb);
    }

    // --- layer u2: hidden @ Wu2 + bu2 -> out (f32) ---
    #pragma unroll
    for (int nt = 0; nt < 8; ++nt) { v8f z = {}; acc[nt] = z; }
    #pragma unroll
    for (int ks = 0; ks < 4; ++ks) {
        v16h a = ld_pair(in + (lane & 15) * INSTRIDE + ks * 32 + kb_a, 0, 16);
        #pragma unroll
        for (int nt = 0; nt < 8; ++nt) {
            v16h b = ld_pair(sWu2t + (nt * 16 + ncol) * WSTRIDE + ks * 32 + kb_b, 0, 16);
            acc[nt] = wmma16(a, b, acc[nt]);
        }
    }
    #pragma unroll
    for (int r = 0; r < 8; ++r) {
        int row = base + mrow + r;
        if (row < NS) {
            #pragma unroll
            for (int nt = 0; nt < 8; ++nt)
                out[(size_t)row * HDIM + nt * 16 + ncol] = acc[nt][r] + sbu2[nt * 16 + ncol];
        }
    }
}

extern "C" void kernel_launch(void* const* d_in, const int* in_sizes, int n_in,
                              void* d_out, int out_size, void* d_ws, size_t ws_size,
                              hipStream_t stream) {
    const float* node_emb = (const float*)d_in[0];
    const float* node_pos = (const float*)d_in[1];
    const float* grid_pos = (const float*)d_in[2];
    const int*   eidx     = (const int*)d_in[3];
    const float* frames   = (const float*)d_in[4];
    const int*   batch    = (const int*)d_in[5];
    const float* We1 = (const float*)d_in[6];
    const float* be1 = (const float*)d_in[7];
    const float* We2 = (const float*)d_in[8];
    const float* be2 = (const float*)d_in[9];
    const float* Wm1 = (const float*)d_in[10];
    const float* bm1 = (const float*)d_in[11];
    const float* Wm2 = (const float*)d_in[12];
    const float* bm2 = (const float*)d_in[13];
    const float* Wu1 = (const float*)d_in[14];
    const float* bu1 = (const float*)d_in[15];
    const float* Wu2 = (const float*)d_in[16];
    const float* bu2 = (const float*)d_in[17];

    const int E  = in_sizes[3] / 2;
    const int Nn = in_sizes[5];
    const int G  = in_sizes[2] / 3;
    const int NS = Nn + G;

    float* seg_sum = (float*)d_ws;
    float* seg_cnt = seg_sum + (size_t)NS * HDIM;
    size_t zero_bytes = ((size_t)NS * HDIM + (size_t)NS) * sizeof(float);
    hipMemsetAsync(d_ws, 0, zero_bytes, stream);

    hipFuncSetAttribute(reinterpret_cast<const void*>(mpnn_edge_kernel),
                        hipFuncAttributeMaxDynamicSharedMemorySize, SMEM_A);
    hipFuncSetAttribute(reinterpret_cast<const void*>(mpnn_update_kernel),
                        hipFuncAttributeMaxDynamicSharedMemorySize, SMEM_B);

    const int blocksA = (E + 127) / 128;
    mpnn_edge_kernel<<<blocksA, 256, SMEM_A, stream>>>(
        node_emb, node_pos, grid_pos, eidx, frames, batch,
        We1, be1, We2, be2, Wm1, bm1, Wm2, bm2,
        seg_sum, seg_cnt, E, Nn);

    const int blocksB = (NS + 127) / 128;
    mpnn_update_kernel<<<blocksB, 256, SMEM_B, stream>>>(
        seg_sum, seg_cnt, Wu1, bu1, Wu2, bu2, (float*)d_out, NS);
}
